// SegmentedAttention_53042846106004
// MI455X (gfx1250) — matrix operand order repaired and checked by reading, not compiled, not run on hardware
//
#include <hip/hip_runtime.h>
#include <hip/hip_bf16.h>

typedef unsigned short u16;
typedef __attribute__((ext_vector_type(16))) __bf16 v16bf;
typedef __attribute__((ext_vector_type(8)))  float  v8f;

#define H_   16
#define DH_  64
#define SEG_ 512
#define P_   16
#define KROWS 544   /* P + SEG padded to multiple of 32 */
#define NTOK 16384  /* B*N */
#define D_   1024

__device__ __forceinline__ u16 f2bf(float f) {
  unsigned int u = __float_as_uint(f);
  unsigned int r = (u + 0x7FFFu + ((u >> 16) & 1u)) >> 16;
  return (u16)r;
}
__device__ __forceinline__ float bf2f(u16 h) {
  return __uint_as_float(((unsigned int)h) << 16);
}
__device__ __forceinline__ v8f v8f_zero() {
  v8f z = {0.f,0.f,0.f,0.f,0.f,0.f,0.f,0.f};
  return z;
}

// A operand (16x32 bf16, M x K), src row-major with leading dim ld.
__device__ __forceinline__ v16bf load_mat_a(const u16* p, int ld, int m0, int k0, int lane) {
  const int row = m0 + (lane & 15);
  const int kb  = k0 + ((lane & 16) ? 8 : 0);
  union { v16bf v; uint4 q[2]; } u;
  u.q[0] = *(const uint4*)(p + (size_t)row * ld + kb);
  u.q[1] = *(const uint4*)(p + (size_t)row * ld + kb + 16);
  return u.v;
}

// B operand (32x16 bf16, K x N), src stored transposed (N x K row-major, ld).
__device__ __forceinline__ v16bf load_mat_bt(const u16* p, int ld, int n0, int k0, int lane) {
  const int col = n0 + (lane & 15);
  const int kb  = k0 + ((lane & 16) ? 8 : 0);
  union { v16bf v; uint4 q[2]; } u;
  u.q[0] = *(const uint4*)(p + (size_t)col * ld + kb);
  u.q[1] = *(const uint4*)(p + (size_t)col * ld + kb + 16);
  return u.v;
}

__device__ __forceinline__ v8f wmma_bf16(v16bf a, v16bf b, v8f c) {
  return __builtin_amdgcn_wmma_f32_16x16x32_bf16(false, a, false, b, (short)0, c, false, false);
}

// Async DMA: 16B global -> LDS per lane, tracked by ASYNCcnt (CDNA5).
// VDST = LDS byte address = low 32 bits of the generic shared pointer.
__device__ __forceinline__ void async_g2l_b128(const void* g, void* lds_generic) {
  asm volatile("global_load_async_to_lds_b128 %0, %1, off"
               :: "v"((unsigned)(unsigned long long)lds_generic), "v"(g)
               : "memory");
}
__device__ __forceinline__ void wait_asynccnt0() {
  asm volatile("s_wait_asynccnt 0x0" ::: "memory");
}

// ---------------- RMSNorm + cast to bf16 (one block per token) ----------------
__global__ __launch_bounds__(256) void k_rmsnorm(const float* __restrict__ seq,
                                                 const float* __restrict__ w,
                                                 u16* __restrict__ xh) {
  __shared__ float red[256];
  const int t = blockIdx.x, tid = threadIdx.x;
  const float4 v  = ((const float4*)(seq + (size_t)t * D_))[tid];
  red[tid] = v.x*v.x + v.y*v.y + v.z*v.z + v.w*v.w;
  __syncthreads();
  for (int s = 128; s > 0; s >>= 1) {
    if (tid < s) red[tid] += red[tid + s];
    __syncthreads();
  }
  const float r = rsqrtf(red[0] * (1.0f / D_) + 1.1920929e-07f);
  const float4 wv = ((const float4*)w)[tid];
  u16* o = xh + (size_t)t * D_ + tid * 4;
  o[0] = f2bf(v.x * r * wv.x);
  o[1] = f2bf(v.y * r * wv.y);
  o[2] = f2bf(v.z * r * wv.z);
  o[3] = f2bf(v.w * r * wv.w);
}

// ---------------- transpose + cast weights: Wt[n*K+k] = bf16(W[k*N+n]) ----------------
__global__ __launch_bounds__(256) void k_transpose_bf16(const float* __restrict__ W,
                                                        u16* __restrict__ Wt,
                                                        int K, int N) {
  long long i = (long long)blockIdx.x * 256 + threadIdx.x;
  if (i >= (long long)K * N) return;
  int n = (int)(i / K), k = (int)(i % K);
  Wt[i] = f2bf(W[(size_t)k * N + n]);
}

// ---------------- bf16 WMMA GEMM with async double-buffered LDS staging ----------------
// C(MxN) = A(MxK) * Bt(NxK)^T.  Block tile 64x256, 8 waves (2 M-groups x 4 N-groups),
// wave tile 32x64 (8 accumulators).  K-panel of 64 staged per buffer:
//   A: 64x64 bf16 (8 KB), B: 256x64 bf16 (32 KB); double buffered (80 KB LDS,
//   4 blocks resident per 320 KB WGP).  16 WMMAs per barrier.
#define KP 64
#define GA_ELEMS (64 * KP)
#define GB_ELEMS (256 * KP)
#define GBUF_ELEMS (GA_ELEMS + GB_ELEMS)

__device__ __forceinline__ void stage_panel(const u16* __restrict__ A,
                                            const u16* __restrict__ Bt,
                                            int K, int m0, int n0, int k0,
                                            u16* smA, u16* smB, int tid) {
#pragma unroll
  for (int t = 0; t < 2; ++t) {                 // A: 64 rows x 8 chunks of 16B
    const int idx = tid + t * 256;
    const int row = idx >> 3, ch = (idx & 7) * 8;
    async_g2l_b128(A + (size_t)(m0 + row) * K + k0 + ch, smA + row * KP + ch);
  }
#pragma unroll
  for (int t = 0; t < 8; ++t) {                 // B: 256 rows x 8 chunks of 16B
    const int idx = tid + t * 256;
    const int row = idx >> 3, ch = (idx & 7) * 8;
    async_g2l_b128(Bt + (size_t)(n0 + row) * K + k0 + ch, smB + row * KP + ch);
  }
}

template <bool OUT_BF16>
__global__ __launch_bounds__(256) void k_gemm(const u16* __restrict__ A,
                                              const u16* __restrict__ Bt,
                                              void* __restrict__ Cout,
                                              int N, int K) {
  __shared__ __align__(16) u16 smem[2][GBUF_ELEMS];
  const int tid  = threadIdx.x;
  const int lane = tid & 31;
  const int wid  = tid >> 5;
  const int m0 = blockIdx.x * 64;
  const int n0 = blockIdx.y * 256;
  const int mrow0 = (wid >> 2) * 32;        // wave's 32 rows within the 64-row tile
  const int ncol0 = (wid & 3) * 64;         // wave's 64 cols within the 256-col tile

  v8f acc[2][4];
#pragma unroll
  for (int mi = 0; mi < 2; ++mi)
#pragma unroll
    for (int j = 0; j < 4; ++j) acc[mi][j] = v8f_zero();

  stage_panel(A, Bt, K, m0, n0, 0, smem[0], smem[0] + GA_ELEMS, tid);

  int cur = 0;
  for (int k0 = 0; k0 < K; k0 += KP) {
    wait_asynccnt0();                       // own async writes for buffer `cur` arrived
    __syncthreads();                        // everyone's arrived; prior reads of `nxt` done
    if (k0 + KP < K)
      stage_panel(A, Bt, K, m0, n0, k0 + KP, smem[cur ^ 1], smem[cur ^ 1] + GA_ELEMS, tid);

    const u16* smA = smem[cur];
    const u16* smB = smem[cur] + GA_ELEMS;
#pragma unroll
    for (int ks = 0; ks < KP; ks += 32) {
      const v16bf a0 = load_mat_a(smA, KP, mrow0,      ks, lane);
      const v16bf a1 = load_mat_a(smA, KP, mrow0 + 16, ks, lane);
#pragma unroll
      for (int j = 0; j < 4; ++j) {
        const v16bf b = load_mat_bt(smB, KP, ncol0 + j * 16, ks, lane);
        acc[0][j] = wmma_bf16(a0, b, acc[0][j]);
        acc[1][j] = wmma_bf16(a1, b, acc[1][j]);
      }
    }
    cur ^= 1;
  }

  const int hi8 = (lane & 16) ? 8 : 0;
  const int cb  = n0 + ncol0 + (lane & 15);
#pragma unroll
  for (int mi = 0; mi < 2; ++mi) {
    union { v8f v; float f[8]; } u[4];
#pragma unroll
    for (int j = 0; j < 4; ++j) u[j].v = acc[mi][j];
#pragma unroll
    for (int j = 0; j < 4; ++j)
#pragma unroll
      for (int r = 0; r < 8; ++r) {
        const size_t idx = (size_t)(m0 + mrow0 + mi * 16 + r + hi8) * N + cb + j * 16;
        if (OUT_BF16) ((u16*)Cout)[idx] = f2bf(u[j].f[r]);
        else          ((float*)Cout)[idx] = u[j].f[r];
      }
  }
}

// ---------------- persistent-memory fill + pad zeroing ----------------
__global__ __launch_bounds__(256) void k_pmfill(const float* __restrict__ pm,
                                                u16* __restrict__ Kb,
                                                u16* __restrict__ Vt) {
  int i = blockIdx.x * 256 + threadIdx.x;          // 32*16*16*64
  if (i >= 32 * H_ * P_ * DH_) return;
  const int d  = i & 63;
  const int p  = (i >> 6) & 15;
  const int h  = (i >> 10) & 15;
  const int bw = i >> 14;
  const int bwh = bw * H_ + h;
  Kb[((size_t)bwh * KROWS + p) * DH_ + d]       = f2bf(pm[((0*H_ + h)*P_ + p)*DH_ + d]);
  Vt[((size_t)bwh * DH_ + d) * KROWS + p]       = f2bf(pm[((1*H_ + h)*P_ + p)*DH_ + d]);
  Kb[((size_t)bwh * KROWS + (P_ + SEG_) + p) * DH_ + d] = 0;
  Vt[((size_t)bwh * DH_ + d) * KROWS + (P_ + SEG_) + p] = 0;
}

// ---------------- RoPE + segment repack (Q row-major, K row-major, V transposed) ----------------
__global__ __launch_bounds__(256) void k_rope(const u16* __restrict__ qkv,
                                              u16* __restrict__ Qb,
                                              u16* __restrict__ Kb,
                                              u16* __restrict__ Vt) {
  int i = blockIdx.x * 256 + threadIdx.x;          // NTOK * H * (DH/2)
  if (i >= NTOK * H_ * (DH_ / 2)) return;
  const int ii = i & 31;
  const int h  = (i >> 5) & 15;
  const int t  = i >> 9;
  const int pos   = t & 8191;
  const int batch = t >> 13;
  const float freq = __expf(-(float)(2 * ii) * 0.14391156847f);  // ln(10000)/64
  const float ang = (float)pos * freq;
  float s, c;
  __sincosf(ang, &s, &c);
  const size_t base = (size_t)t * (3 * H_ * DH_) + h * DH_ + 2 * ii;
  const float q1 = bf2f(qkv[base]),        q2 = bf2f(qkv[base + 1]);
  const float k1 = bf2f(qkv[base + 1024]), k2 = bf2f(qkv[base + 1025]);
  const float v1 = bf2f(qkv[base + 2048]), v2 = bf2f(qkv[base + 2049]);
  const int bwh = (batch * 16 + (pos >> 9)) * H_ + h;
  const int sl  = pos & (SEG_ - 1);
  size_t qo = ((size_t)bwh * SEG_ + sl) * DH_ + 2 * ii;
  Qb[qo]     = f2bf(q1 * c - q2 * s);
  Qb[qo + 1] = f2bf(q2 * c + q1 * s);
  size_t ko = ((size_t)bwh * KROWS + P_ + sl) * DH_ + 2 * ii;
  Kb[ko]     = f2bf(k1 * c - k2 * s);
  Kb[ko + 1] = f2bf(k2 * c + k1 * s);
  size_t vo = ((size_t)bwh * DH_ + 2 * ii) * KROWS + P_ + sl;
  Vt[vo]         = f2bf(v1);
  Vt[vo + KROWS] = f2bf(v2);
}

// ---------------- flash attention per (bw,h): 8 waves x 16 q-rows per block ----------------
// Softmax kept in log2 domain: scores scaled once by DH^-0.5 * log2(e), exp2f thereafter.
__global__ __launch_bounds__(256) void k_attn(const u16* __restrict__ Qb,
                                              const u16* __restrict__ Kb,
                                              const u16* __restrict__ Vt,
                                              u16* __restrict__ att) {
  __shared__ __align__(16) u16 smem[8 * 16 * 32];   // per-wave 16x32 bf16 P tile
  const int lane = threadIdx.x & 31;
  const int wid  = threadIdx.x >> 5;
  const int bwh  = blockIdx.x >> 2;                 // 0..511
  const int qc   = blockIdx.x & 3;
  const int bw   = bwh >> 4;
  const int h    = bwh & 15;
  const int qi0  = qc * 128 + wid * 16;

  const u16* Q  = Qb + (size_t)bwh * SEG_ * DH_;
  const u16* Kp = Kb + (size_t)bwh * KROWS * DH_;
  const u16* Vp = Vt + (size_t)bwh * DH_ * KROWS;
  u16* sm = smem + wid * (16 * 32);

  const v16bf aq0 = load_mat_a(Q, DH_, qi0, 0,  lane);
  const v16bf aq1 = load_mat_a(Q, DH_, qi0, 32, lane);

  const int hi8 = (lane & 16) ? 8 : 0;
  const int cl  = lane & 15;
  const float NEG = -__builtin_inff();
  const float SCL = 0.125f * 1.4426950408889634f;   // DH^-0.5 * log2(e)

  float m[8], l[8];
#pragma unroll
  for (int r = 0; r < 8; ++r) { m[r] = NEG; l[r] = 0.f; }
  v8f o0 = v8f_zero(), o1 = v8f_zero(), o2 = v8f_zero(), o3 = v8f_zero();

  const int lim = min(P_ + SEG_, qi0 + 32);         // keys needed: j <= qi0+15+P
  for (int kj = 0; kj < lim; kj += 32) {
    v8f s0 = v8f_zero(), s1 = v8f_zero();
    s0 = wmma_bf16(aq0, load_mat_bt(Kp, DH_, kj,      0,  lane), s0);
    s0 = wmma_bf16(aq1, load_mat_bt(Kp, DH_, kj,      32, lane), s0);
    s1 = wmma_bf16(aq0, load_mat_bt(Kp, DH_, kj + 16, 0,  lane), s1);
    s1 = wmma_bf16(aq1, load_mat_bt(Kp, DH_, kj + 16, 32, lane), s1);

    union { v8f v; float f[8]; } u0, u1, w0, w1, w2, w3;
    u0.v = s0; u1.v = s1;
    float alpha[8];
#pragma unroll
    for (int r = 0; r < 8; ++r) {
      const int row = qi0 + r + hi8;
      const int c0  = kj + cl;
      const int c1  = kj + 16 + cl;
      float a0 = u0.f[r] * SCL;                      // log2-domain score
      float a1 = u1.f[r] * SCL;
      if (c0 > row + P_ || c0 >= P_ + SEG_) a0 = NEG;
      if (c1 > row + P_ || c1 >= P_ + SEG_) a1 = NEG;
      float x = fmaxf(a0, a1);
      x = fmaxf(x, __shfl_xor(x, 1, 32));
      x = fmaxf(x, __shfl_xor(x, 2, 32));
      x = fmaxf(x, __shfl_xor(x, 4, 32));
      x = fmaxf(x, __shfl_xor(x, 8, 32));
      const float mn = fmaxf(m[r], x);
      alpha[r] = exp2f(m[r] - mn);
      const float e0 = exp2f(a0 - mn);
      const float e1 = exp2f(a1 - mn);
      float srow = e0 + e1;
      srow += __shfl_xor(srow, 1, 32);
      srow += __shfl_xor(srow, 2, 32);
      srow += __shfl_xor(srow, 4, 32);
      srow += __shfl_xor(srow, 8, 32);
      l[r] = l[r] * alpha[r] + srow;
      m[r] = mn;
      sm[(r + hi8) * 32 + cl]      = f2bf(e0);       // P tile, D-layout -> row-major LDS
      sm[(r + hi8) * 32 + 16 + cl] = f2bf(e1);
    }
    w0.v = o0; w1.v = o1; w2.v = o2; w3.v = o3;
#pragma unroll
    for (int r = 0; r < 8; ++r) {
      w0.f[r] *= alpha[r]; w1.f[r] *= alpha[r];
      w2.f[r] *= alpha[r]; w3.f[r] *= alpha[r];
    }
    o0 = w0.v; o1 = w1.v; o2 = w2.v; o3 = w3.v;

    asm volatile("s_wait_dscnt 0x0" ::: "memory");   // wave-local LDS store->load fence
    const v16bf pA = load_mat_a(sm, 32, 0, 0, lane); // re-read P in A-operand layout

    o0 = wmma_bf16(pA, load_mat_bt(Vp, KROWS,  0, kj, lane), o0);
    o1 = wmma_bf16(pA, load_mat_bt(Vp, KROWS, 16, kj, lane), o1);
    o2 = wmma_bf16(pA, load_mat_bt(Vp, KROWS, 32, kj, lane), o2);
    o3 = wmma_bf16(pA, load_mat_bt(Vp, KROWS, 48, kj, lane), o3);
    asm volatile("s_wait_dscnt 0x0" ::: "memory");   // LDS reads done before next overwrite
  }

  const int batch = bw >> 4, win = bw & 15;
  const size_t rowbase = (size_t)batch * 8192 + win * SEG_ + qi0;
  union { v8f v; float f[8]; } w0, w1, w2, w3;
  w0.v = o0; w1.v = o1; w2.v = o2; w3.v = o3;
#pragma unroll
  for (int r = 0; r < 8; ++r) {
    const float inv = 1.0f / l[r];
    const size_t row = rowbase + r + hi8;
    u16* dst = att + row * (H_ * DH_) + h * DH_ + cl;
    dst[0]  = f2bf(w0.f[r] * inv);
    dst[16] = f2bf(w1.f[r] * inv);
    dst[32] = f2bf(w2.f[r] * inv);
    dst[48] = f2bf(w3.f[r] * inv);
  }
}

extern "C" void kernel_launch(void* const* d_in, const int* in_sizes, int n_in,
                              void* d_out, int out_size, void* d_ws, size_t ws_size,
                              hipStream_t stream) {
  (void)in_sizes; (void)n_in; (void)out_size; (void)ws_size;
  const float* seq  = (const float*)d_in[0];
  const float* wn   = (const float*)d_in[1];
  const float* Wqkv = (const float*)d_in[2];
  const float* Wout = (const float*)d_in[3];
  const float* pm   = (const float*)d_in[4];
  float* out = (float*)d_out;

  size_t off = 0;
  auto take = [&](size_t bytes) -> char* {
    char* p = (char*)d_ws + off;
    off += (bytes + 255) & ~(size_t)255;
    return p;
  };
  u16* xh    = (u16*)take((size_t)NTOK * D_ * 2);
  u16* wqkvT = (u16*)take((size_t)3 * D_ * D_ * 2);
  u16* woutT = (u16*)take((size_t)D_ * D_ * 2);
  u16* qkv   = (u16*)take((size_t)NTOK * 3 * D_ * 2);
  u16* Qbuf  = (u16*)take((size_t)512 * SEG_ * DH_ * 2);
  u16* Kbuf  = (u16*)take((size_t)512 * KROWS * DH_ * 2);
  u16* Vtb   = (u16*)take((size_t)512 * DH_ * KROWS * 2);
  u16* att   = (u16*)take((size_t)NTOK * D_ * 2);

  k_rmsnorm<<<NTOK, 256, 0, stream>>>(seq, wn, xh);
  k_transpose_bf16<<<(3 * D_ * D_ + 255) / 256, 256, 0, stream>>>(Wqkv, wqkvT, D_, 3 * D_);
  k_transpose_bf16<<<(D_ * D_ + 255) / 256, 256, 0, stream>>>(Wout, woutT, D_, D_);
  k_gemm<true><<<dim3(NTOK / 64, (3 * D_) / 256), 256, 0, stream>>>(xh, wqkvT, qkv, 3 * D_, D_);
  k_pmfill<<<(32 * H_ * P_ * DH_ + 255) / 256, 256, 0, stream>>>(pm, Kbuf, Vtb);
  k_rope<<<(NTOK * H_ * (DH_ / 2)) / 256, 256, 0, stream>>>(qkv, Qbuf, Kbuf, Vtb);
  k_attn<<<512 * 4, 256, 0, stream>>>(Qbuf, Kbuf, Vtb, att);
  k_gemm<false><<<dim3(NTOK / 64, D_ / 256), 256, 0, stream>>>(att, woutT, out, D_, D_);
}
